// EGNNLayer_51951924412709
// MI455X (gfx1250) — matrix operand order repaired and checked by reading, not compiled, not run on hardware
//
#include <hip/hip_runtime.h>
#include <hip/hip_bf16.h>
#include <math.h>

typedef __attribute__((ext_vector_type(16))) _Float16 v16h;
typedef __attribute__((ext_vector_type(8)))  _Float16 v8h;
typedef __attribute__((ext_vector_type(8)))  float    v8f;

namespace {
constexpr int Bb = 2, Nn = 256, Pp = 768, Hh = 64, Ee = 32;
constexpr int AS = 160;            // Abuf row stride (f16 elems)
constexpr float FINF = 1e9f;

// ---- dynamic LDS layout (bytes) ----
constexpr int WMSG1_OFF = 0;                     // [160][64] f16 (swizzled frag order)
constexpr int WMSG2_OFF = WMSG1_OFF + 160*64*2;  // [64][64]
constexpr int WATT1_OFF = WMSG2_OFF + 64*64*2;   // [96][64] (66 padded)
constexpr int WROT1_OFF = WATT1_OFF + 96*64*2;   // [96][64] (68 padded)
constexpr int WTOR1_OFF = WROT1_OFF + 96*64*2;   // [96][64] (78 padded)
constexpr int WTRN1_OFF = WTOR1_OFF + 96*64*2;   // [64][64]
constexpr int ATT2_OFF  = WTRN1_OFF + 64*64*2;   // [64]    f32
constexpr int ROT2_OFF  = ATT2_OFF + 64*4;       // [64*4]  f32
constexpr int TOR2_OFF  = ROT2_OFF + 256*4;      // [64*7]  f32
constexpr int TRN2_OFF  = TOR2_OFF + 448*4;      // [64]    f32
constexpr int BIAS_OFF  = TRN2_OFF + 64*4;       // [6*64]  f32
constexpr int HI_OFF    = BIAS_OFF + 384*4;      // [64]    f16  h_i
constexpr int MSGSUM_OFF= HI_OFF + 64*2;         // [64]    f32
constexpr int FEATH_OFF = MSGSUM_OFF + 64*4;     // [64]    f32
constexpr int STATE_OFF = FEATH_OFF + 64*4;      // [128][20] f32 softmax states
constexpr int ABUF_OFF  = STATE_OFF + 8*16*20*4; // [8][16*160] f16 per-wave A tiles
constexpr int HBUF_OFF  = ABUF_OFF + 8*16*AS*2;  // [8][16*64]  f16 per-wave hidden
constexpr int SMEM_BYTES= HBUF_OFF + 8*16*64*2;  // = 146816 B (fits 320KB WGP LDS)
} // namespace

__device__ __forceinline__ v8f wmma32(v16h a, v16h b, v8f c) {
  return __builtin_amdgcn_wmma_f32_16x16x32_f16(false, a, false, b, (short)0, c,
                                                false, false);
}

// A fragment 16x32 f16 from LDS tile buf[16][ld], chunk starting at column k0.
// ISA layout: lane m=lane&15 is row; half=lane>>4 selects K 0-7/16-23 vs 8-15/24-31.
// Two contiguous 16B reads per lane.
__device__ __forceinline__ v16h load_A(const _Float16* buf, int ld, int k0, int lane) {
  int m = lane & 15, half = lane >> 4;
  const _Float16* p = buf + m * ld + k0 + half * 8;
  v8h lo = *(const v8h*)(p);
  v8h hi = *(const v8h*)(p + 16);
  v16h r;
#pragma unroll
  for (int e = 0; e < 8; ++e) { r[e] = lo[e]; r[8 + e] = hi[e]; }
  return r;
}

// B fragment from pre-swizzled weight store: chunk (kt,nt) -> 32 lanes x 16 f16,
// contiguous 32B per lane.
__device__ __forceinline__ v16h load_Bswz(const _Float16* w, int kt, int nt, int lane) {
  return *(const v16h*)(w + (((kt * 4 + nt) * 32) + lane) * 16);
}

// D fragment (f32 16x16) -> f16 LDS tile buf[16][ld], column tile n0.
__device__ __forceinline__ void storeD(_Float16* buf, int ld, int n0, v8f d, int lane) {
  int n = n0 + (lane & 15), half = lane >> 4;
#pragma unroll
  for (int r = 0; r < 8; ++r) buf[(r + half * 8) * ld + n] = (_Float16)d[r];
}

// copy f32 global weight [K][64] into f16 LDS in B-fragment-swizzled order,
// zero-padding K up to Kpad.
__device__ __forceinline__ void copy_w_swz(_Float16* dst, const float* src,
                                           int K, int Kpad, int tid) {
  for (int idx = tid; idx < Kpad * 64; idx += 256) {
    int k = idx >> 6, n = idx & 63;
    int sidx = ((((k >> 5) * 4 + (n >> 4)) * 32) + (((k & 8) << 1) + (n & 15))) * 16 + ((k & 7) + ((k & 16) >> 1));
    dst[sidx] = (k < K) ? (_Float16)src[k * 64 + n] : (_Float16)0.0f;
  }
}

__device__ __forceinline__ void qmul(const float* q, const float* p, float* r) {
  r[0] = q[0]*p[0] - q[1]*p[1] - q[2]*p[2] - q[3]*p[3];
  r[1] = q[0]*p[1] + q[1]*p[0] + q[2]*p[3] - q[3]*p[2];
  r[2] = q[0]*p[2] - q[1]*p[3] + q[2]*p[0] + q[3]*p[1];
  r[3] = q[0]*p[3] + q[1]*p[2] - q[2]*p[1] + q[3]*p[0];
}

__global__ __launch_bounds__(256)
void egnn_fused(const float* __restrict__ pep_quats, const float* __restrict__ pep_trans,
                const float* __restrict__ pep_torsions, const float* __restrict__ pep_feat,
                const float* __restrict__ pep_edge, const unsigned char* __restrict__ pep_mask,
                const float* __restrict__ pocket_feat, const float* __restrict__ pocket_quats,
                const float* __restrict__ pocket_trans, const unsigned char* __restrict__ pocket_mask,
                const float* __restrict__ msg_W1, const float* __restrict__ msg_b1,
                const float* __restrict__ msg_W2, const float* __restrict__ msg_b2,
                const float* __restrict__ att_W1, const float* __restrict__ att_b1,
                const float* __restrict__ att_W2, const float* __restrict__ att_b2,
                const float* __restrict__ feat_W1, const float* __restrict__ feat_b1,
                const float* __restrict__ feat_W2, const float* __restrict__ feat_b2,
                const float* __restrict__ rot_W1, const float* __restrict__ rot_b1,
                const float* __restrict__ rot_W2, const float* __restrict__ rot_b2,
                const float* __restrict__ tor_W1, const float* __restrict__ tor_b1,
                const float* __restrict__ tor_W2, const float* __restrict__ tor_b2,
                const float* __restrict__ trn_W1, const float* __restrict__ trn_b1,
                const float* __restrict__ trn_W2, const float* __restrict__ trn_b2,
                float* __restrict__ out)
{
  extern __shared__ char smem[];
  _Float16* sWmsg1 = (_Float16*)(smem + WMSG1_OFF);
  _Float16* sWmsg2 = (_Float16*)(smem + WMSG2_OFF);
  _Float16* sWatt1 = (_Float16*)(smem + WATT1_OFF);
  _Float16* sWrot1 = (_Float16*)(smem + WROT1_OFF);
  _Float16* sWtor1 = (_Float16*)(smem + WTOR1_OFF);
  _Float16* sWtrn1 = (_Float16*)(smem + WTRN1_OFF);
  float* sAtt2   = (float*)(smem + ATT2_OFF);
  float* sRot2   = (float*)(smem + ROT2_OFF);
  float* sTor2   = (float*)(smem + TOR2_OFF);
  float* sTrn2   = (float*)(smem + TRN2_OFF);
  float* sBias   = (float*)(smem + BIAS_OFF);
  _Float16* sHi  = (_Float16*)(smem + HI_OFF);
  float* sMsgSum = (float*)(smem + MSGSUM_OFF);
  float* sFeatH  = (float*)(smem + FEATH_OFF);
  float* sState  = (float*)(smem + STATE_OFF);

  const int tid  = threadIdx.x;
  const int wave = tid >> 5;
  const int lane = tid & 31;
  const int b = blockIdx.x / Nn;
  const int i = blockIdx.x % Nn;
  const size_t node = (size_t)b * Nn + i;

  // ---- phase 0: weights -> LDS (f16, swizzled) ----
  copy_w_swz(sWmsg1, msg_W1, 160, 160, tid);
  copy_w_swz(sWmsg2, msg_W2,  64,  64, tid);
  copy_w_swz(sWatt1, att_W1,  66,  96, tid);
  copy_w_swz(sWrot1, rot_W1,  68,  96, tid);
  copy_w_swz(sWtor1, tor_W1,  78,  96, tid);
  copy_w_swz(sWtrn1, trn_W1,  64,  64, tid);
  for (int k = tid; k < 448; k += 256) sTor2[k] = tor_W2[k];
  for (int k = tid; k < 256; k += 256) sRot2[k] = rot_W2[k];
  if (tid < 64) {
    sAtt2[tid] = att_W2[tid];
    sTrn2[tid] = trn_W2[tid];
    sBias[tid]       = msg_b1[tid];
    sBias[64  + tid] = msg_b2[tid];
    sBias[128 + tid] = att_b1[tid];
    sBias[192 + tid] = rot_b1[tid];
    sBias[256 + tid] = tor_b1[tid];
    sBias[320 + tid] = trn_b1[tid];
    sHi[tid] = (_Float16)pep_feat[node * Hh + tid];
    sMsgSum[tid] = 0.0f;
  }

  // ---- phase 1: per-thread node-i state ----
  float qi[4], ti[3], ft[14];
#pragma unroll
  for (int q = 0; q < 4; ++q) qi[q] = pep_quats[node * 4 + q];
#pragma unroll
  for (int d = 0; d < 3; ++d) ti[d] = pep_trans[node * 3 + d];
#pragma unroll
  for (int k = 0; k < 14; ++k) ft[k] = pep_torsions[node * 14 + k];
  const int mask_i = pep_mask[node] != 0;
  const float b2_att = att_b2[0];
  const float b2_trn = trn_b2[0];
  float b2_rot[4], b2_tor[7];
#pragma unroll
  for (int c = 0; c < 4; ++c) b2_rot[c] = rot_b2[c];
#pragma unroll
  for (int c = 0; c < 7; ++c) b2_tor[c] = tor_b2[c];

  __syncthreads();

  _Float16* Ab = (_Float16*)(smem + ABUF_OFF) + wave * 16 * AS;
  _Float16* Hb = (_Float16*)(smem + HBUF_OFF) + wave * 16 * 64;

  // online softmax state (per lane<16, one edge per tile)
  float run_m = -3.0e38f, run_s = 0.0f, cnt = 0.0f;
  float accv[14];
#pragma unroll
  for (int k = 0; k < 14; ++k) accv[k] = 0.0f;
  float msum[4] = {0.f, 0.f, 0.f, 0.f};

  // ---- phase 2: j loop, 8 waves x 16 edges per iteration ----
  for (int it = 0; it < 8; ++it) {
    const int j0 = it * 128 + wave * 16;

    // stage X = [h_i | h_j | e] into Abuf (16 x 160)
    for (int r = 0; r < 16; ++r) {
      const int j = j0 + r;
      const float* hj = (j < Nn) ? pep_feat + ((size_t)b * Nn + j) * Hh
                                 : pocket_feat + ((size_t)b * Pp + (j - Nn)) * Hh;
      Ab[r * AS + lane]       = sHi[lane];
      Ab[r * AS + 32 + lane]  = sHi[32 + lane];
      Ab[r * AS + 64 + lane]  = (_Float16)hj[lane];
      Ab[r * AS + 96 + lane]  = (_Float16)hj[32 + lane];
      Ab[r * AS + 128 + lane] = (j < Nn)
          ? (_Float16)pep_edge[(((size_t)b * Nn + i) * Nn + j) * Ee + lane]
          : (_Float16)0.0f;
    }

    // GEMM1 msg: [16x160] @ [160x64]
    v16h xa[5];
#pragma unroll
    for (int kt = 0; kt < 5; ++kt) xa[kt] = load_A(Ab, AS, kt * 32, lane);
#pragma unroll
    for (int nt = 0; nt < 4; ++nt) {
      v8f acc = {0.f,0.f,0.f,0.f,0.f,0.f,0.f,0.f};
#pragma unroll
      for (int kt = 0; kt < 5; ++kt)
        acc = wmma32(xa[kt], load_Bswz(sWmsg1, kt, nt, lane), acc);
      float bb = sBias[nt * 16 + (lane & 15)];
#pragma unroll
      for (int r = 0; r < 8; ++r) acc[r] = fmaxf(acc[r] + bb, 0.0f);
      storeD(Hb, 64, nt * 16, acc, lane);
    }

    // GEMM2 msg: [16x64] @ [64x64] -> msg; accumulate msg_sum; stage msg->Abuf[0:64]
    v16h h0 = load_A(Hb, 64, 0, lane), h1 = load_A(Hb, 64, 32, lane);
#pragma unroll
    for (int nt = 0; nt < 4; ++nt) {
      v8f acc = {0.f,0.f,0.f,0.f,0.f,0.f,0.f,0.f};
      acc = wmma32(h0, load_Bswz(sWmsg2, 0, nt, lane), acc);
      acc = wmma32(h1, load_Bswz(sWmsg2, 1, nt, lane), acc);
      float bb = sBias[64 + nt * 16 + (lane & 15)];
      float ps = 0.0f;
#pragma unroll
      for (int r = 0; r < 8; ++r) { acc[r] += bb; ps += acc[r]; }
      msum[nt] += ps;
      storeD(Ab, AS, nt * 16, acc, lane);
    }

    // zero extra columns 64..95 (whole wave)
    for (int idx = lane; idx < 512; idx += 32)
      Ab[(idx >> 5) * AS + 64 + (idx & 31)] = (_Float16)0.0f;

    // per-edge neighbour data + att extras (lanes 0..15)
    const int j = j0 + (lane & 15);
    const bool elane = lane < 16;
    float qj[4] = {0,0,0,0}, tj[3] = {0,0,0}, inq[4] = {0,0,0,0};
    int validv = 0;
    if (elane) {
      const float* qs = (j < Nn) ? pep_quats + ((size_t)b * Nn + j) * 4
                                 : pocket_quats + ((size_t)b * Pp + (j - Nn)) * 4;
      const float* ts = (j < Nn) ? pep_trans + ((size_t)b * Nn + j) * 3
                                 : pocket_trans + ((size_t)b * Pp + (j - Nn)) * 3;
#pragma unroll
      for (int q = 0; q < 4; ++q) qj[q] = qs[q];
#pragma unroll
      for (int d = 0; d < 3; ++d) tj[d] = ts[d];
      int mj = (j < Nn) ? (pep_mask[(size_t)b * Nn + j] != 0 && j != i)
                        : (pocket_mask[(size_t)b * Pp + (j - Nn)] != 0);
      validv = mask_i && mj;
      float d2 = 0.0f;
#pragma unroll
      for (int d = 0; d < 3; ++d) { float dd = ti[d] - tj[d]; d2 += dd * dd; }
      float qd = 0.0f;
#pragma unroll
      for (int q = 0; q < 4; ++q) qd += qi[q] * qj[q];
      Ab[lane * AS + 64] = (_Float16)(-d2);
      Ab[lane * AS + 65] = (_Float16)(qd * qd);
    }

    // att GEMM1: [16x96] @ [96x64]
    v16h a0 = load_A(Ab, AS, 0, lane);
    v16h a1 = load_A(Ab, AS, 32, lane);
    v16h a2 = load_A(Ab, AS, 64, lane);
#pragma unroll
    for (int nt = 0; nt < 4; ++nt) {
      v8f acc = {0.f,0.f,0.f,0.f,0.f,0.f,0.f,0.f};
      acc = wmma32(a0, load_Bswz(sWatt1, 0, nt, lane), acc);
      acc = wmma32(a1, load_Bswz(sWatt1, 1, nt, lane), acc);
      acc = wmma32(a2, load_Bswz(sWatt1, 2, nt, lane), acc);
      float bb = sBias[128 + nt * 16 + (lane & 15)];
#pragma unroll
      for (int r = 0; r < 8; ++r) acc[r] = fmaxf(acc[r] + bb, 0.0f);
      storeD(Hb, 64, nt * 16, acc, lane);
    }

    // att layer2 (64->1) + masked logit; rot extras (local_q)
    float logitv = 0.0f;
    if (elane) {
      float s = b2_att;
      for (int k = 0; k < 64; ++k) s += (float)Hb[lane * 64 + k] * sAtt2[k];
      logitv = s - (validv ? 0.0f : FINF);
      float n2 = qj[0]*qj[0] + qj[1]*qj[1] + qj[2]*qj[2] + qj[3]*qj[3];
      float inv = 1.0f / n2;
      inq[0] = qj[0]*inv; inq[1] = -qj[1]*inv; inq[2] = -qj[2]*inv; inq[3] = -qj[3]*inv;
      float t1[4], lq[4];
      qmul(qi, qj, t1);
      qmul(inq, t1, lq);
#pragma unroll
      for (int q = 0; q < 4; ++q) Ab[lane * AS + 64 + q] = (_Float16)lq[q];
    }

    // rot GEMM1
    a2 = load_A(Ab, AS, 64, lane);
#pragma unroll
    for (int nt = 0; nt < 4; ++nt) {
      v8f acc = {0.f,0.f,0.f,0.f,0.f,0.f,0.f,0.f};
      acc = wmma32(a0, load_Bswz(sWrot1, 0, nt, lane), acc);
      acc = wmma32(a1, load_Bswz(sWrot1, 1, nt, lane), acc);
      acc = wmma32(a2, load_Bswz(sWrot1, 2, nt, lane), acc);
      float bb = sBias[192 + nt * 16 + (lane & 15)];
#pragma unroll
      for (int r = 0; r < 8; ++r) acc[r] = fmaxf(acc[r] + bb, 0.0f);
      storeD(Hb, 64, nt * 16, acc, lane);
    }

    // rot layer2 (64->4), sigmoid, gdq = q_j * ldq * q_j^-1 ; tor extras (flat_t)
    float gdq[4] = {0,0,0,0};
    if (elane) {
      float ldq[4];
#pragma unroll
      for (int c = 0; c < 4; ++c) {
        float s = b2_rot[c];
        for (int k = 0; k < 64; ++k) s += (float)Hb[lane * 64 + k] * sRot2[k * 4 + c];
        ldq[c] = 1.0f / (1.0f + expf(-s));
      }
      float t2[4];
      qmul(ldq, inq, t2);
      qmul(qj, t2, gdq);
#pragma unroll
      for (int k = 0; k < 14; ++k) Ab[lane * AS + 64 + k] = (_Float16)ft[k];
    }

    // tor GEMM1
    a2 = load_A(Ab, AS, 64, lane);
#pragma unroll
    for (int nt = 0; nt < 4; ++nt) {
      v8f acc = {0.f,0.f,0.f,0.f,0.f,0.f,0.f,0.f};
      acc = wmma32(a0, load_Bswz(sWtor1, 0, nt, lane), acc);
      acc = wmma32(a1, load_Bswz(sWtor1, 1, nt, lane), acc);
      acc = wmma32(a2, load_Bswz(sWtor1, 2, nt, lane), acc);
      float bb = sBias[256 + nt * 16 + (lane & 15)];
#pragma unroll
      for (int r = 0; r < 8; ++r) acc[r] = fmaxf(acc[r] + bb, 0.0f);
      storeD(Hb, 64, nt * 16, acc, lane);
    }

    // tor layer2 (64->7)
    float mda[7] = {0,0,0,0,0,0,0};
    if (elane) {
#pragma unroll
      for (int c = 0; c < 7; ++c) {
        float s = b2_tor[c];
        for (int k = 0; k < 64; ++k) s += (float)Hb[lane * 64 + k] * sTor2[k * 7 + c];
        mda[c] = s;
      }
    }

    // trn GEMM1 (input = msg only -> reuse a0,a1)
#pragma unroll
    for (int nt = 0; nt < 4; ++nt) {
      v8f acc = {0.f,0.f,0.f,0.f,0.f,0.f,0.f,0.f};
      acc = wmma32(a0, load_Bswz(sWtrn1, 0, nt, lane), acc);
      acc = wmma32(a1, load_Bswz(sWtrn1, 1, nt, lane), acc);
      float bb = sBias[320 + nt * 16 + (lane & 15)];
#pragma unroll
      for (int r = 0; r < 8; ++r) acc[r] = fmaxf(acc[r] + bb, 0.0f);
      storeD(Hb, 64, nt * 16, acc, lane);
    }

    // trn layer2 (64->1) + online-softmax update
    if (elane) {
      float s = b2_trn;
      for (int k = 0; k < 64; ++k) s += (float)Hb[lane * 64 + k] * sTrn2[k];
      float mr[3];
#pragma unroll
      for (int d = 0; d < 3; ++d) mr[d] = s * (ti[d] - tj[d]);

      float nm = fmaxf(run_m, logitv);
      float f1 = expf(run_m - nm);
      float p  = expf(logitv - nm);
      run_s = run_s * f1 + p;
#pragma unroll
      for (int q = 0; q < 4; ++q) accv[q]      = accv[q]      * f1 + p * gdq[q];
#pragma unroll
      for (int c = 0; c < 7; ++c) accv[4 + c]  = accv[4 + c]  * f1 + p * mda[c];
#pragma unroll
      for (int d = 0; d < 3; ++d) accv[11 + d] = accv[11 + d] * f1 + p * mr[d];
      run_m = nm;
      cnt += validv ? 1.0f : 0.0f;
    }
  } // j loop

  // ---- phase 3: reductions across lanes/waves ----
#pragma unroll
  for (int nt = 0; nt < 4; ++nt)
    atomicAdd(&sMsgSum[nt * 16 + (lane & 15)], msum[nt]);
  if (lane < 16) {
    float* st = sState + (wave * 16 + lane) * 20;
    st[0] = run_m; st[1] = run_s;
#pragma unroll
    for (int k = 0; k < 14; ++k) st[2 + k] = accv[k];
    st[16] = cnt;
  }
  __syncthreads();

  float* out_q = out;
  float* out_x = out + Bb * Nn * 4;
  float* out_t = out + Bb * Nn * (4 + 3);
  float* out_o = out + Bb * Nn * (4 + 3 + 14);

  if (tid == 0) {
    float m = -3.0e38f, s = 0.0f, c = 0.0f, A[14];
#pragma unroll
    for (int k = 0; k < 14; ++k) A[k] = 0.0f;
    for (int st = 0; st < 128; ++st) {
      const float* p = sState + st * 20;
      float nm = fmaxf(m, p[0]);
      float f1 = expf(m - nm), f2 = expf(p[0] - nm);
      s = s * f1 + p[1] * f2;
#pragma unroll
      for (int k = 0; k < 14; ++k) A[k] = A[k] * f1 + p[2 + k] * f2;
      c += p[16];
      m = nm;
    }
    float invs = 1.0f / s;
    float gd[4];
    if (c > 0.5f) {
#pragma unroll
      for (int q = 0; q < 4; ++q) gd[q] = A[q] * invs;
    } else {
      gd[0] = 1.0f; gd[1] = gd[2] = gd[3] = 0.0f;
    }
    float gn = 1.0f / sqrtf(gd[0]*gd[0] + gd[1]*gd[1] + gd[2]*gd[2] + gd[3]*gd[3]);
#pragma unroll
    for (int q = 0; q < 4; ++q) gd[q] *= gn;
    float uq[4];
    qmul(gd, qi, uq);
    float un = 1.0f / sqrtf(uq[0]*uq[0] + uq[1]*uq[1] + uq[2]*uq[2] + uq[3]*uq[3]);
#pragma unroll
    for (int q = 0; q < 4; ++q) out_q[node * 4 + q] = uq[q] * un;
#pragma unroll
    for (int k = 0; k < 7; ++k) {
      float da = A[4 + k] * invs;
      float dsn = sinf(da), dcn = cosf(da);
      float tsn = ft[2 * k], tcn = ft[2 * k + 1];
      out_t[node * 14 + 2 * k]     = dsn * tcn + dcn * tsn;
      out_t[node * 14 + 2 * k + 1] = dcn * tcn - dsn * tsn;
    }
#pragma unroll
    for (int d = 0; d < 3; ++d) out_x[node * 3 + d] = ti[d] + A[11 + d] * invs;
  }

  // feat MLP (128->64->64) once per node, threads 64..127
  if (tid >= 64 && tid < 128) {
    int t = tid - 64;
    float s = feat_b1[t];
    const float* hi = pep_feat + node * Hh;
    for (int k = 0; k < 64; ++k) s += hi[k] * feat_W1[k * 64 + t];
    for (int k = 0; k < 64; ++k) s += sMsgSum[k] * feat_W1[(64 + k) * 64 + t];
    sFeatH[t] = fmaxf(s, 0.0f);
  }
  __syncthreads();
  if (tid >= 64 && tid < 128) {
    int t = tid - 64;
    float s = feat_b2[t];
    for (int k = 0; k < 64; ++k) s += sFeatH[k] * feat_W2[k * 64 + t];
    out_o[node * 64 + t] = s;
  }
}

extern "C" void kernel_launch(void* const* d_in, const int* in_sizes, int n_in,
                              void* d_out, int out_size, void* d_ws, size_t ws_size,
                              hipStream_t stream) {
  (void)in_sizes; (void)n_in; (void)d_ws; (void)ws_size; (void)out_size;
  const float* pep_quats      = (const float*)d_in[0];
  const float* pep_trans      = (const float*)d_in[1];
  const float* pep_torsions   = (const float*)d_in[2];
  const float* pep_feat       = (const float*)d_in[3];
  const float* pep_edge       = (const float*)d_in[4];
  const unsigned char* pep_mask    = (const unsigned char*)d_in[5];
  const float* pocket_feat    = (const float*)d_in[6];
  const float* pocket_quats   = (const float*)d_in[7];
  const float* pocket_trans   = (const float*)d_in[8];
  const unsigned char* pocket_mask = (const unsigned char*)d_in[9];
  const float* msg_W1 = (const float*)d_in[10]; const float* msg_b1 = (const float*)d_in[11];
  const float* msg_W2 = (const float*)d_in[12]; const float* msg_b2 = (const float*)d_in[13];
  const float* att_W1 = (const float*)d_in[14]; const float* att_b1 = (const float*)d_in[15];
  const float* att_W2 = (const float*)d_in[16]; const float* att_b2 = (const float*)d_in[17];
  const float* feat_W1 = (const float*)d_in[18]; const float* feat_b1 = (const float*)d_in[19];
  const float* feat_W2 = (const float*)d_in[20]; const float* feat_b2 = (const float*)d_in[21];
  const float* rot_W1 = (const float*)d_in[22]; const float* rot_b1 = (const float*)d_in[23];
  const float* rot_W2 = (const float*)d_in[24]; const float* rot_b2 = (const float*)d_in[25];
  const float* tor_W1 = (const float*)d_in[26]; const float* tor_b1 = (const float*)d_in[27];
  const float* tor_W2 = (const float*)d_in[28]; const float* tor_b2 = (const float*)d_in[29];
  const float* trn_W1 = (const float*)d_in[30]; const float* trn_b1 = (const float*)d_in[31];
  const float* trn_W2 = (const float*)d_in[32]; const float* trn_b2 = (const float*)d_in[33];
  float* out = (float*)d_out;

  (void)hipFuncSetAttribute((const void*)egnn_fused,
                            hipFuncAttributeMaxDynamicSharedMemorySize,
                            (int)SMEM_BYTES);
  egnn_fused<<<dim3(Bb * Nn), dim3(256), SMEM_BYTES, stream>>>(
      pep_quats, pep_trans, pep_torsions, pep_feat, pep_edge, pep_mask,
      pocket_feat, pocket_quats, pocket_trans, pocket_mask,
      msg_W1, msg_b1, msg_W2, msg_b2,
      att_W1, att_b1, att_W2, att_b2,
      feat_W1, feat_b1, feat_W2, feat_b2,
      rot_W1, rot_b1, rot_W2, rot_b2,
      tor_W1, tor_b1, tor_W2, tor_b2,
      trn_W1, trn_b1, trn_W2, trn_b2,
      out);
}
